// Encoder_67173288509869
// MI455X (gfx1250) — matrix operand order repaired and checked by reading, not compiled, not run on hardware
//
#include <hip/hip_runtime.h>
#include <hip/hip_bf16.h>

// ---------------------------------------------------------------------------
// Problem constants (match reference)
// ---------------------------------------------------------------------------
#define BB      8
#define LL      2048
#define DMODEL  512
#define DTYPE   32
#define NPLANE  (BB * LL)          // 16384 events total
#define REGP    5.0f
#define BETA0   0.4f
#define BETA1   0.3f
#define EPSV    1e-6f
// ln(10000)/256  (div_term exponent step)
#define LOG1E4_OVER_256 0.035977892084121704f

typedef __attribute__((ext_vector_type(16))) _Float16 v16h;
typedef __attribute__((ext_vector_type(8)))  float    v8f;
typedef __attribute__((ext_vector_type(4)))  float    v4f;   // native vector: OK for NT builtins

// ---------------------------------------------------------------------------
// Kernel A: projections via WMMA.
//   pa = te @ w_l[:32], pb = te @ w_l[32:], ga = te @ w_g[:32], gb = te @ w_g[32:]
// One wave handles 16 events: A = 16x32 f16 tile of gathered type embeddings,
// B = 32x16 f16 with weight vectors in columns 0..3 (rest zero).
// D[m, 0..3] = {pa, pb, ga, gb} for event row0+m. Stored SoA into d_ws planes.
// Emits v_wmma_f32_16x16x32_f16.
// ---------------------------------------------------------------------------
__global__ void __launch_bounds__(256)
proj_wmma_kernel(const int* __restrict__ etype,
                 const float* __restrict__ type_emb,
                 const float* __restrict__ w_l,
                 const float* __restrict__ w_g,
                 float* __restrict__ proj)   // 4 planes of NPLANE floats
{
    const int lane = threadIdx.x & 31;
    const int wave = (blockIdx.x * blockDim.x + threadIdx.x) >> 5;
    const int row0 = wave * 16;            // first of 16 events for this wave
    const int m    = lane & 15;
    const int half = lane >> 4;            // 0: lanes 0-15, 1: lanes 16-31

    // ---- B fragment: 32x16 f16, same K order as A: element e ->
    //      K = (e < 8 ? 0 : 16) + half*8 + (e & 7).
    v16h bfrag = {};
    const int n = lane & 15;               // output column this lane owns
    if (n < 4) {
        const float* w = (n == 0) ? w_l
                       : (n == 1) ? (w_l + DTYPE)
                       : (n == 2) ? w_g
                       :            (w_g + DTYPE);
        const int kb = half * 8;
#pragma unroll
        for (int e = 0; e < 16; ++e)
            bfrag[e] = (_Float16)w[((e < 8) ? 0 : 16) + kb + (e & 7)];
    }

    // ---- A fragment: 16x32 f16, ISA layout (16-bit A 16x32):
    //   lanes 0-15:  v0..v3 -> K 0..7,  v4..v7 -> K 16..23
    //   lanes 16-31: v0..v3 -> K 8..15, v4..v7 -> K 24..31
    const int   t  = etype[row0 + m];
    const float* te = type_emb + t * DTYPE;
    v16h afrag;
#pragma unroll
    for (int v = 0; v < 8; ++v) {
        const int kb = ((v < 4) ? 0 : 16) + half * 8 + (v & 3) * 2;
        afrag[2 * v]     = (_Float16)te[kb];
        afrag[2 * v + 1] = (_Float16)te[kb + 1];
    }

    v8f c = {};
    c = __builtin_amdgcn_wmma_f32_16x16x32_f16(
            /*neg_a=*/false, afrag, /*neg_b=*/false, bfrag,
            /*c_mod=*/(short)0, c, /*reuse_a=*/false, /*reuse_b=*/false);

    // ---- D layout: lane holds column N = lane&15; c[r] is row M = r + 8*half.
    if (n < 4) {
        float* plane = proj + n * NPLANE + row0 + half * 8;
#pragma unroll
        for (int r = 0; r < 8; ++r)
            plane[r] = c[r];
    }
}

// ---------------------------------------------------------------------------
// Kernel B: hidden_vector[b,l, :] = [sin(arc+phi), cos(arc+phi), te_row]
// One block per event (b*L + l), 256 threads: thread k does sin/cos pair k,
// threads 0..31 also copy the 32-wide type embedding. Output is streamed once
// and never re-read -> non-temporal stores.
// ---------------------------------------------------------------------------
__global__ void __launch_bounds__(256)
hidden_kernel(const float* __restrict__ etime,
              const int* __restrict__ etype,
              const float* __restrict__ Wt_pos,
              const float* __restrict__ type_emb,
              float* __restrict__ outh)
{
    const int bl = blockIdx.x;                 // b*L + l
    const int l  = bl & (LL - 1);
    const int k  = threadIdx.x;                // 0..255
    float* row = outh + (size_t)bl * (DMODEL + DTYPE);

    const float t    = etime[bl];
    const float phi  = t * Wt_pos[k];
    const float divt = __expf(-(float)k * LOG1E4_OVER_256);
    const float arc  = fmaf((float)l, divt, phi);

    float sv, cv;
    __sincosf(arc, &sv, &cv);
    __builtin_nontemporal_store(sv, row + k);
    __builtin_nontemporal_store(cv, row + k + DMODEL / 2);

    if (k < DTYPE) {
        const float te = type_emb[etype[bl] * DTYPE + k];
        __builtin_nontemporal_store(te, row + DMODEL + k);
    }
}

// ---------------------------------------------------------------------------
// Kernel C: pairwise scores + t_diff. One block per (b,i) row; 512 threads,
// 4 consecutive j per thread (v4f loads, b128 NT stores). Fully-masked
// quads (j0 >= i) skip all transcendental work; whole waves above the
// diagonal are EXECZ-skipped, halving VALU cost.
// 256 MB of write-once output -> non-temporal stores so the hot per-row
// arrays (event_time + 4 proj planes, ~320 KB total, reused 2048x) are not
// evicted from cache by the output stream. HBM traffic ~= the writes.
// ---------------------------------------------------------------------------
__device__ __forceinline__ float score_elem(float d, float pa, float pb_i,
                                            float ga, float gb_i,
                                            float bl, float bg)
{
    // lengthscale = softplus(pa+pb+b_l) + eps   (numerically stable softplus)
    float x  = pa + pb_i + bl;
    float sp = fmaxf(x, 0.0f) + __logf(1.0f + __expf(-fabsf(x)));
    float lv = sp + EPSV;
    float rl = __builtin_amdgcn_rcpf(lv);
    // gate = sigmoid(REG * (ga+gb+b_g))
    float g  = __builtin_amdgcn_rcpf(1.0f + __expf(-REGP * (ga + gb_i + bg)));
    float dr  = d * rl;
    float kse = __expf(-0.5f * dr * dr);
    float kex = __expf(-dr);
    return g * (BETA0 * kse + BETA1 * kex);
}

__global__ void __launch_bounds__(512)
pair_kernel(const float* __restrict__ etime,
            const float* __restrict__ proj,
            const float* __restrict__ pbl,   // scalar b_l
            const float* __restrict__ pbg,   // scalar b_g
            float* __restrict__ scores,
            float* __restrict__ tdiff)
{
    const int bi = blockIdx.x;               // b*L + i
    const int b  = bi >> 11;
    const int i  = bi & (LL - 1);
    const int rb = b * LL;

    const float* tb = etime + rb;
    const float ti   = tb[i];
    const float pb_i = proj[1 * NPLANE + rb + i];
    const float gb_i = proj[3 * NPLANE + rb + i];
    const float bl   = pbl[0];
    const float bg   = pbg[0];

    const int j0 = threadIdx.x << 2;

    const v4f tj = *(const v4f*)(tb + j0);
    v4f td;
    td.x = fabsf(tj.x - ti);
    td.y = fabsf(tj.y - ti);
    td.z = fabsf(tj.z - ti);
    td.w = fabsf(tj.w - ti);

    v4f sc = {0.f, 0.f, 0.f, 0.f};
    if (j0 < i) {   // at least one unmasked element in this quad
        const v4f pa = *(const v4f*)(proj + 0 * NPLANE + rb + j0);
        const v4f ga = *(const v4f*)(proj + 2 * NPLANE + rb + j0);
        sc.x =                    score_elem(td.x, pa.x, pb_i, ga.x, gb_i, bl, bg);
        sc.y = (j0 + 1 < i) ? score_elem(td.y, pa.y, pb_i, ga.y, gb_i, bl, bg) : 0.f;
        sc.z = (j0 + 2 < i) ? score_elem(td.z, pa.z, pb_i, ga.z, gb_i, bl, bg) : 0.f;
        sc.w = (j0 + 3 < i) ? score_elem(td.w, pa.w, pb_i, ga.w, gb_i, bl, bg) : 0.f;
    }

    const size_t ob = (size_t)bi * LL + j0;
    __builtin_nontemporal_store(sc, (v4f*)(scores + ob));  // b128 TH=NT
    __builtin_nontemporal_store(td, (v4f*)(tdiff  + ob));  // b128 TH=NT
}

// ---------------------------------------------------------------------------
// Launch
// ---------------------------------------------------------------------------
extern "C" void kernel_launch(void* const* d_in, const int* in_sizes, int n_in,
                              void* d_out, int out_size, void* d_ws, size_t ws_size,
                              hipStream_t stream)
{
    const int*   etype = (const int*)  d_in[0];   // event_type   [B,L] i32
    const float* etime = (const float*)d_in[1];   // event_time   [B,L]
    // d_in[2] arrival_times: unused by reference
    const float* Wt    = (const float*)d_in[3];   // Wt_pos       [256]
    const float* temb  = (const float*)d_in[4];   // type_emb     [21,32]
    const float* w_l   = (const float*)d_in[5];   // [64]
    const float* b_l   = (const float*)d_in[6];   // scalar
    const float* w_g   = (const float*)d_in[7];   // [64]
    const float* b_g   = (const float*)d_in[8];   // scalar

    float* out    = (float*)d_out;
    float* scores = out;                                            // [B,L,L]
    float* hidden = scores + (size_t)BB * LL * LL;                  // [B,L,544]
    float* tdiff  = hidden + (size_t)BB * LL * (DMODEL + DTYPE);    // [B,L,L]

    float* proj = (float*)d_ws;   // 4 SoA planes {pa,pb,ga,gb} of NPLANE floats

    // A: 16384 events / 16 per wave = 1024 waves; 8 waves per 256-thr block
    proj_wmma_kernel<<<NPLANE / 16 / 8, 256, 0, stream>>>(etype, temb, w_l, w_g, proj);

    // B: one block per event
    hidden_kernel<<<BB * LL, 256, 0, stream>>>(etime, etype, Wt, temb, hidden);

    // C: one block per (b,i) row, 4 j per thread
    pair_kernel<<<BB * LL, LL / 4, 0, stream>>>(etime, proj, b_l, b_g, scores, tdiff);
}
